// SVTISDPA_25555055411788
// MI455X (gfx1250) — matrix operand order repaired and checked by reading, not compiled, not run on hardware
//
#include <hip/hip_runtime.h>
#include <hip/hip_bf16.h>
#include <math.h>

// Problem constants (match reference)
#define H_ 8
#define E_ 256
#define V_ 1024
#define M_ 64
#define B_ 512
#define QKV_ 6144   // 3*E*H
#define HE_ 2048    // H*E

typedef __attribute__((ext_vector_type(16))) _Float16 v16h;
typedef __attribute__((ext_vector_type(8)))  _Float16 v8h;
typedef __attribute__((ext_vector_type(8)))  float    v8f;

// ---------------- positional encoding (64 x 256) ----------------
__global__ void k_pos(float* __restrict__ pe) {
  int idx = blockIdx.x * blockDim.x + threadIdx.x;   // 64*128
  if (idx >= M_ * (E_ / 2)) return;
  int m = idx >> 7, k = idx & 127;
  float f  = __expf(-(float)k * (logf(10000.0f) / 128.0f)); // freq[k] = 10000^(-k/128)
  float ph = (float)m * f;
  pe[m * E_ + 2 * k]     = sinf(ph);
  pe[m * E_ + 2 * k + 1] = cosf(ph);
}

// ---------------- f32 -> f16 ----------------
__global__ void k_cvt16(_Float16* __restrict__ dst, const float* __restrict__ src, int n) {
  int i = blockIdx.x * blockDim.x + threadIdx.x;
  if (i < n) dst[i] = (_Float16)src[i];
}

// ---------------- WMMA fragment helpers ----------------
// A fragment (16x32 f16, contiguous layout): per lane two 16B chunks at +0,+8
__device__ inline v16h load_afrag(const _Float16* p) {
  v8h lo = *(const v8h*)p;
  v8h hi = *(const v8h*)(p + 8);
  return __builtin_shufflevector(lo, hi, 0,1,2,3,4,5,6,7,8,9,10,11,12,13,14,15);
}
__device__ inline v8f wmma16(v16h a, v16h b, v8f c) {
  return __builtin_amdgcn_wmma_f32_16x16x32_f16(false, a, false, b, (short)0, c,
                                                false, false);
}

// ---------------- batched NT GEMM via WMMA, 32x32 tile / wave ----------------
// C[b][m][n] = scale * sum_k A[b][m][k] * B[b][n][k]  (+ C if accumulate)
// One wave computes a 32x32 tile (2x2 WMMA sub-tiles, fragment reuse).
// Requires: M,N multiples of 32; K multiple of 32.
__global__ void k_gemm_nt(const _Float16* __restrict__ A, long long sA, int lda,
                          const _Float16* __restrict__ B, long long sB, int ldb,
                          float* __restrict__ C, long long sC, int ldc,
                          int K, float scale, int accumulate) {
  int lane = threadIdx.x & 31;
  int n0 = blockIdx.x * 32;
  int m0 = blockIdx.y * 32;
  A += (long long)blockIdx.z * sA;
  B += (long long)blockIdx.z * sB;
  C += (long long)blockIdx.z * sC;

  int r  = lane & 15;            // row (A) / col (B) within 16-wide sub-tile
  int ka = (lane >> 4) * 16;     // A half-wave K offset (contiguous 16 per half-wave)
  int kb = (lane >> 4) * 16;     // B half-wave K offset (contiguous 16 per half-wave)
  const _Float16* pa0 = A + (size_t)(m0 + r) * lda + ka;
  const _Float16* pa1 = pa0 + (size_t)16 * lda;
  const _Float16* pb0 = B + (size_t)(n0 + r) * ldb + kb;
  const _Float16* pb1 = pb0 + (size_t)16 * ldb;

  v8f acc00 = {}, acc01 = {}, acc10 = {}, acc11 = {};
  for (int k = 0; k < K; k += 32) {
    v16h a0 = load_afrag(pa0 + k);
    v16h a1 = load_afrag(pa1 + k);
    v16h b0 = *(const v16h*)(pb0 + k);
    v16h b1 = *(const v16h*)(pb1 + k);
    acc00 = wmma16(a0, b0, acc00);
    acc01 = wmma16(a0, b1, acc01);
    acc10 = wmma16(a1, b0, acc10);
    acc11 = wmma16(a1, b1, acc11);
  }

  // C/D layout: vgpr i -> M = i (lanes 0-15) / i+8 (lanes 16-31); N = lane&15
  int mb = (lane >> 4) * 8;
  v8f accs[2][2] = {{acc00, acc01}, {acc10, acc11}};
#pragma unroll
  for (int dm = 0; dm < 2; ++dm)
#pragma unroll
    for (int dn = 0; dn < 2; ++dn) {
      int n = n0 + dn * 16 + r;
#pragma unroll
      for (int i = 0; i < 8; ++i) {
        size_t off = (size_t)(m0 + dm * 16 + mb + i) * ldc + n;
        float v = accs[dm][dn][i] * scale;
        if (accumulate) v += C[off];
        C[off] = v;
      }
    }
}

// ---------------- char_qk[:, :, PAD] = -1e9 ----------------
__global__ void k_fix(float* __restrict__ cqk) {
  int i = blockIdx.x * blockDim.x + threadIdx.x;  // H*V
  if (i < H_ * V_) cqk[(size_t)i * V_] = -1.0e9f;
}

// ---------------- gather + softmax + masked pooling ----------------
// grid (H, B), block 64. pooled[b,h,j] = sum_i mask_i*softmax_row_i(j) / n_words
__global__ void k_attn(const int* __restrict__ cc, const float* __restrict__ cqk,
                       const float* __restrict__ cpqk, const float* __restrict__ pqk,
                       float* __restrict__ pooled) {
  int h = blockIdx.x, b = blockIdx.y, t = threadIdx.x;
  __shared__ int   cs[M_];
  __shared__ float a_s[M_][M_ + 1];
  __shared__ float rowm[M_], rowsc[M_];

  cs[t] = cc[b * M_ + t];
  __syncthreads();

  int ci = cs[t];
  const float* qkrow = cqk  + ((size_t)h << 20) + (size_t)ci * V_;
  const float* cprow = cpqk + (size_t)h * (V_ * M_) + (size_t)ci * M_;
  const float* prow  = pqk  + (size_t)(h * M_ + t) * M_;

  float mx = -3.4e38f;
#pragma unroll 4
  for (int j = 0; j < M_; ++j) {
    float s = qkrow[cs[j]] + cprow[j] + prow[j];
    a_s[t][j] = s;
    mx = fmaxf(mx, s);
  }
  float se = 0.0f;
#pragma unroll 4
  for (int j = 0; j < M_; ++j) se += __expf(a_s[t][j] - mx);
  rowm[t]  = mx;
  rowsc[t] = (ci != 0 ? 1.0f : 0.0f) / se;
  __syncthreads();

  float nw = 0.0f;
  for (int i = 0; i < M_; ++i) nw += (cs[i] != 0) ? 1.0f : 0.0f;

  float pj = 0.0f;
#pragma unroll 4
  for (int i = 0; i < M_; ++i) pj += rowsc[i] * __expf(a_s[i][t] - rowm[i]);
  pooled[((size_t)b * H_ + h) * M_ + t] = pj / nw;
}

// ---------------- out1[b, h*E+e] = sum_m pooled*(cv[c[b,m],h,e]+pv[m,h,e]) ----------------
__global__ void k_wv(const int* __restrict__ cc, const float* __restrict__ pooled,
                     const float* __restrict__ cqkv, const float* __restrict__ pqkv,
                     float* __restrict__ out1) {
  int h = blockIdx.x, b = blockIdx.y, e = threadIdx.x;  // 256
  __shared__ int   cs[M_];
  __shared__ float ps[M_];
  if (e < M_) {
    cs[e] = cc[b * M_ + e];
    ps[e] = pooled[((size_t)b * H_ + h) * M_ + e];
  }
  __syncthreads();
  const float* cv = cqkv + h * 768 + 512 + e;
  const float* pv = pqkv + h * 768 + 512 + e;
  float acc = 0.0f;
#pragma unroll 4
  for (int m = 0; m < M_; ++m)
    acc += ps[m] * (cv[(size_t)cs[m] * QKV_] + pv[(size_t)m * QKV_]);
  out1[(size_t)b * HE_ + h * E_ + e] = acc;
}

// ---------------- fused bias + LN x3 chain ----------------
__device__ inline float block_sum(float v, float* red, int t) {
  red[t] = v; __syncthreads();
  for (int s = 128; s > 0; s >>= 1) { if (t < s) red[t] += red[t + s]; __syncthreads(); }
  float r = red[0]; __syncthreads();
  return r;
}
__device__ inline float lnorm(float x, float* red, int t, float eps) {
  float m = block_sum(x, red, t) * (1.0f / E_);
  float d = x - m;
  float v = block_sum(d * d, red, t) * (1.0f / E_);
  return d * rsqrtf(v + eps);
}

__global__ void k_ln3(const float* __restrict__ x, const float* __restrict__ fb,
                      const float* g1, const float* b1, const float* g2, const float* b2,
                      const float* g3, const float* b3, float* __restrict__ out) {
  __shared__ float red[E_];
  int row = blockIdx.x, t = threadIdx.x;
  float v = x[(size_t)row * E_ + t] + fb[t];
  float y = lnorm(v, red, t, 1e-5f) * g1[t] + b1[t];
  float z = lnorm(y, red, t, 1e-6f) * g2[t] + b2[t] + y;
  float o = lnorm(z, red, t, 1e-5f) * g3[t] + b3[t];
  out[(size_t)row * E_ + t] = o;
}

// ---------------- launcher ----------------
extern "C" void kernel_launch(void* const* d_in, const int* in_sizes, int n_in,
                              void* d_out, int out_size, void* d_ws, size_t ws_size,
                              hipStream_t stream) {
  (void)in_sizes; (void)n_in; (void)out_size; (void)ws_size;
  const int*   cc   = (const int*)d_in[0];
  const float* emb  = (const float*)d_in[1];
  const float* qkvw = (const float*)d_in[2];
  const float* fcw  = (const float*)d_in[3];
  const float* fcb  = (const float*)d_in[4];
  const float* g1 = (const float*)d_in[5];  const float* b1 = (const float*)d_in[6];
  const float* g2 = (const float*)d_in[7];  const float* b2 = (const float*)d_in[8];
  const float* g3 = (const float*)d_in[9];  const float* b3 = (const float*)d_in[10];
  // d_in[11] = max_len (== 64, compile-time)

  char* w = (char*)d_ws;
  auto alloc = [&](size_t bytes) -> char* {
    char* p = w; w += (bytes + 255) & ~(size_t)255; return p;
  };
  float*    pe     = (float*)   alloc((size_t)M_ * E_ * 4);
  _Float16* emb16  = (_Float16*)alloc((size_t)V_ * E_ * 2);
  _Float16* qkv16  = (_Float16*)alloc((size_t)QKV_ * E_ * 2);
  _Float16* pe16   = (_Float16*)alloc((size_t)M_ * E_ * 2);
  _Float16* fc16   = (_Float16*)alloc((size_t)E_ * HE_ * 2);
  float*    cqkv   = (float*)   alloc((size_t)V_ * QKV_ * 4);
  float*    pqkv   = (float*)   alloc((size_t)M_ * QKV_ * 4);
  _Float16* cqkv16 = (_Float16*)alloc((size_t)V_ * QKV_ * 2);
  _Float16* pqkv16 = (_Float16*)alloc((size_t)M_ * QKV_ * 2);
  float*    cqk    = (float*)   alloc((size_t)H_ * V_ * V_ * 4);
  float*    pqk    = (float*)   alloc((size_t)H_ * M_ * M_ * 4);
  float*    cpqk   = (float*)   alloc((size_t)H_ * V_ * M_ * 4);
  float*    pooled = (float*)   alloc((size_t)B_ * H_ * M_ * 4);
  float*    out1   = (float*)   alloc((size_t)B_ * HE_ * 4);
  _Float16* out116 = (_Float16*)alloc((size_t)B_ * HE_ * 2);
  float*    fcout  = (float*)   alloc((size_t)B_ * E_ * 4);

  const float inv_t = 1.0f / 16.0f;  // 1/sqrt(E)
  dim3 wv(32);

  // positional encoding + f16 operand conversions
  k_pos<<<(M_ * 128 + 255) / 256, 256, 0, stream>>>(pe);
  k_cvt16<<<(V_ * E_ + 255) / 256, 256, 0, stream>>>(emb16, emb, V_ * E_);
  k_cvt16<<<(QKV_ * E_ + 255) / 256, 256, 0, stream>>>(qkv16, qkvw, QKV_ * E_);
  k_cvt16<<<(M_ * E_ + 255) / 256, 256, 0, stream>>>(pe16, pe, M_ * E_);
  k_cvt16<<<(E_ * HE_ + 255) / 256, 256, 0, stream>>>(fc16, fcw, E_ * HE_);

  // qkv projections: cqkv = emb @ qkv_w^T, pqkv = pos @ qkv_w^T
  k_gemm_nt<<<dim3(QKV_ / 32, V_ / 32, 1), wv, 0, stream>>>(
      emb16, 0, E_, qkv16, 0, E_, cqkv, 0, QKV_, E_, 1.0f, 0);
  k_gemm_nt<<<dim3(QKV_ / 32, M_ / 32, 1), wv, 0, stream>>>(
      pe16, 0, E_, qkv16, 0, E_, pqkv, 0, QKV_, E_, 1.0f, 0);
  k_cvt16<<<((int)((size_t)V_ * QKV_) + 255) / 256, 256, 0, stream>>>(cqkv16, cqkv, V_ * QKV_);
  k_cvt16<<<(M_ * QKV_ + 255) / 256, 256, 0, stream>>>(pqkv16, pqkv, M_ * QKV_);

  // char_qk[h] = cq_h @ ck_h^T / 16   (batched over heads via 768-element stride)
  k_gemm_nt<<<dim3(V_ / 32, V_ / 32, H_), wv, 0, stream>>>(
      cqkv16, 768, QKV_, cqkv16 + 256, 768, QKV_, cqk, (long long)V_ * V_, V_, E_, inv_t, 0);
  // pos_qk[h] = pq_h @ pk_h^T / 16
  k_gemm_nt<<<dim3(M_ / 32, M_ / 32, H_), wv, 0, stream>>>(
      pqkv16, 768, QKV_, pqkv16 + 256, 768, QKV_, pqk, (long long)M_ * M_, M_, E_, inv_t, 0);
  // char_pos_qk[h] = (cq_h @ pk_h^T + ck_h @ pq_h^T) / 16
  k_gemm_nt<<<dim3(M_ / 32, V_ / 32, H_), wv, 0, stream>>>(
      cqkv16, 768, QKV_, pqkv16 + 256, 768, QKV_, cpqk, (long long)V_ * M_, M_, E_, inv_t, 0);
  k_gemm_nt<<<dim3(M_ / 32, V_ / 32, H_), wv, 0, stream>>>(
      cqkv16 + 256, 768, QKV_, pqkv16, 768, QKV_, cpqk, (long long)V_ * M_, M_, E_, inv_t, 1);

  // PAD key column -> -1e9, then fused attention + pooling
  k_fix<<<(H_ * V_ + 255) / 256, 256, 0, stream>>>(cqk);
  k_attn<<<dim3(H_, B_), 64, 0, stream>>>(cc, cqk, cpqk, pqk, pooled);

  // pooled . (cv[c]+pv)
  k_wv<<<dim3(H_, B_), E_, 0, stream>>>(cc, pooled, cqkv, pqkv, out1);

  // fc GEMM (512x256x2048) + fused bias/LN chain
  k_cvt16<<<(B_ * HE_ + 255) / 256, 256, 0, stream>>>(out116, out1, B_ * HE_);
  k_gemm_nt<<<dim3(E_ / 32, B_ / 32, 1), wv, 0, stream>>>(
      out116, 0, HE_, fc16, 0, HE_, fcout, 0, E_, HE_, 1.0f, 0);
  k_ln3<<<B_, E_, 0, stream>>>(fcout, fcb, g1, b1, g2, b2, g3, b3, (float*)d_out);
}
